// DiffusionNet_72430328480000
// MI455X (gfx1250) — matrix operand order repaired and checked by reading, not compiled, not run on hardware
//
#include <hip/hip_runtime.h>
#include <hip/hip_bf16.h>
#include <math.h>

// ---------------------------------------------------------------------------
// CDNA5 (gfx1250) DiffusionNet forward.  All dense matmuls run on
// v_wmma_f32_16x16x32_bf16 (f32 storage, bf16 tiles in LDS, f32 accumulate).
// Round 2: 2 M-tiles per wave (B-fragment reuse x2) + software-pipelined
// LDS fragment fetch to overlap ds_load latency with the WMMA pipe.
// ---------------------------------------------------------------------------

typedef __attribute__((ext_vector_type(16))) __bf16 v16bf;
typedef __attribute__((ext_vector_type(8)))  float  v8f;

#define LDP 136   // padded LDS row length (bf16 elements), even -> dword pairs aligned

// ---- WMMA fragment helpers (ISA 7.12.2 layouts, wave32) --------------------

// A-matrix 16x32 bf16: lane L -> row m = L&15; half h=L>>4:
//   VGPR j<4 : K = 8h + 2j,(+1);  VGPR j>=4 : K = 16 + 8h + 2(j-4),(+1)
// LDS tile layout T[m][k] (k contiguous) -> each pair is one aligned dword.
__device__ __forceinline__ v16bf frag_A(const __bf16* T, int mbase, int kb) {
  int lane = threadIdx.x & 31;
  const unsigned* row = (const unsigned*)(T + (mbase + (lane & 15)) * LDP);
  int kk0 = ((lane >> 4) << 3) + (kb << 5);
  union { v16bf v; unsigned u[8]; } r;
#pragma unroll
  for (int j = 0; j < 8; ++j) {
    int kk = kk0 + ((j < 4) ? (j << 1) : (16 + ((j - 4) << 1)));
    r.u[j] = row[kk >> 1];
  }
  return r.v;
}

// B-matrix 32x16 bf16: lane L -> col n = L&15; half h=L>>4:
//   VGPR j<4 : K = 8h + 2j,(+1);  VGPR j>=4 : K = 16 + 8h + 2(j-4),(+1)
// LDS tile layout T[n][k] (k contiguous, i.e. W transposed).
__device__ __forceinline__ v16bf frag_B(const __bf16* T, int ct, int kb) {
  int lane = threadIdx.x & 31;
  const unsigned* row = (const unsigned*)(T + (((ct << 4) + (lane & 15))) * LDP);
  int kbase = ((lane >> 4) << 3) + (kb << 5);
  union { v16bf v; unsigned u[8]; } r;
#pragma unroll
  for (int j = 0; j < 8; ++j) r.u[j] = row[(kbase + ((j < 4) ? (j << 1) : (16 + ((j - 4) << 1)))) >> 1];
  return r.v;
}

__device__ __forceinline__ v8f wmma_bf16(v16bf a, v16bf b, v8f c) {
  return __builtin_amdgcn_wmma_f32_16x16x32_bf16(false, a, false, b,
                                                 (short)0, c, false, false);
}

__device__ __forceinline__ v16bf neg_frag(v16bf a) {
  union { v16bf v; unsigned u[8]; } r; r.v = a;
#pragma unroll
  for (int j = 0; j < 8; ++j) r.u[j] ^= 0x80008000u;
  return r.v;
}

// ---------------------------------------------------------------------------
// Generic 128-wide GEMM: out = act( sum_p A_p @ (s_p * W_p) + bias + resid )
// A_p : [Mb x 128] f32 (row-major, batch stride sA)
// W_p : [128 x 128] f32 (row-major k,c; batch stride sW)
// 256 threads = 8 waves; wave w owns rows {row0+16w..+15} and
// {row0+128+16w..+15} x all 128 cols -> each B fragment feeds 2 WMMAs.
// ---------------------------------------------------------------------------
__global__ __launch_bounds__(256) void gemm128_kernel(
    const float* __restrict__ A0, const float* __restrict__ W0, float s0,
    const float* __restrict__ A1, const float* __restrict__ W1, float s1,
    const float* __restrict__ A2, const float* __restrict__ W2, float s2,
    const float* __restrict__ bias, const float* __restrict__ resid,
    float* __restrict__ out,
    int Mb, long sA, long sW, long sO, int act)
{
  extern __shared__ __align__(16) char smem[];
  __bf16* As = (__bf16*)smem;            // [r][k], 256 rows
  __bf16* Ws = As + 256 * LDP;           // [c][k] (transposed)
  int b    = blockIdx.y;
  int row0 = blockIdx.x * 256;
  int w    = threadIdx.x >> 5;
  int m0   = w << 4;                     // tile 0: LDS rows 0..127
  int m1   = 128 + (w << 4);             // tile 1: LDS rows 128..255
  const float* Ap[3] = {A0, A1, A2};
  const float* Wp[3] = {W0, W1, W2};
  float        sp[3] = {s0, s1, s2};
  v8f acc0[8] = {}, acc1[8] = {};
  for (int p = 0; p < 3; ++p) {
    if (Ap[p] == nullptr) break;
    __syncthreads();
    const float* Ag = Ap[p] + (long)b * sA;
    const float* Wg = Wp[p] + (long)b * sW;
    for (int l = threadIdx.x; l < 256 * 128; l += 256) {
      int r = l >> 7, k = l & 127;
      int row = row0 + r;
      As[r * LDP + k] = (__bf16)((row < Mb) ? Ag[(long)row * 128 + k] : 0.f);
      if (l < 128 * 128)   // W: l>>7 = k, l&127 = c ; store transposed & scaled
        Ws[(l & 127) * LDP + (l >> 7)] = (__bf16)(Wg[l] * sp[p]);
    }
    __syncthreads();
#pragma unroll
    for (int kb = 0; kb < 4; ++kb) {
      v16bf a0 = frag_A(As, m0, kb);
      v16bf a1 = frag_A(As, m1, kb);
      v16bf bcur = frag_B(Ws, 0, kb);
#pragma unroll
      for (int ct = 0; ct < 8; ++ct) {
        v16bf bnext;
        if (ct < 7) bnext = frag_B(Ws, ct + 1, kb);  // prefetch while WMMA runs
        acc0[ct] = wmma_bf16(a0, bcur, acc0[ct]);
        acc1[ct] = wmma_bf16(a1, bcur, acc1[ct]);
        if (ct < 7) bcur = bnext;
      }
    }
  }
  int lane = threadIdx.x & 31;
  float*       outb = out + (long)b * sO;
  const float* resb = resid ? (resid + (long)b * sO) : nullptr;
#pragma unroll
  for (int t = 0; t < 2; ++t) {
    const v8f* acc = t ? acc1 : acc0;
    int mb = t ? m1 : m0;
#pragma unroll
    for (int ct = 0; ct < 8; ++ct) {
      int col = (ct << 4) + (lane & 15);
#pragma unroll
      for (int j = 0; j < 8; ++j) {
        int row = row0 + mb + ((lane >> 4) << 3) + j;
        if (row < Mb) {
          float v = acc[ct][j];
          if (bias) v += bias[col];
          if (resb) v += resb[(long)row * 128 + col];
          if (act == 1) v = fmaxf(v, 0.f);
          outb[(long)row * 128 + col] = v;
        }
      }
    }
  }
}

// ---------------------------------------------------------------------------
// Split-K spectral projection: xspec[b,k,c] += sum_n ev[b,n,k]*x[b,n,c]*m[b,n]
// A (k x n, transposed-in-LDS evecs), B (n x c, transposed-in-LDS x*mass).
// Each workgroup reduces a chunk of n and atomically accumulates 128x128.
// ---------------------------------------------------------------------------
__global__ __launch_bounds__(256) void spec_proj_kernel(
    const float* __restrict__ evecs, const float* __restrict__ x,
    const float* __restrict__ mass, float* __restrict__ xspec,
    int Nn, int chunk)
{
  extern __shared__ __align__(16) char smem[];
  __bf16* Es = (__bf16*)smem;      // [k][n]
  __bf16* Xs = Es + 128 * LDP;     // [c][n]
  int  b    = blockIdx.y;
  long base = (long)b * Nn * 128;
  int  n0   = blockIdx.x * chunk;
  int  nend = n0 + chunk; if (nend > Nn) nend = Nn;
  int  mbase = (threadIdx.x >> 5) << 4;
  v8f acc[8] = {};
  for (int nb = n0; nb < nend; nb += 128) {
    __syncthreads();
    if (nb + 128 < nend)   // warm caches for next sub-tile
      __builtin_prefetch(evecs + base + (long)(nb + 128) * 128, 0, 3);
    for (int l = threadIdx.x; l < 128 * 128; l += 256) {
      int r = l >> 7, k = l & 127;
      int n = nb + r;
      float e = 0.f, xv = 0.f;
      if (n < Nn) {
        e  = evecs[base + (long)n * 128 + k];
        xv = x[base + (long)n * 128 + k] * mass[(long)b * Nn + n];
      }
      Es[k * LDP + r] = (__bf16)e;    // transpose on store
      Xs[k * LDP + r] = (__bf16)xv;
    }
    __syncthreads();
#pragma unroll
    for (int kb = 0; kb < 4; ++kb) {
      v16bf a = frag_A(Es, mbase, kb);
      v16bf bcur = frag_B(Xs, 0, kb);
#pragma unroll
      for (int ct = 0; ct < 8; ++ct) {
        v16bf bnext;
        if (ct < 7) bnext = frag_B(Xs, ct + 1, kb);
        acc[ct] = wmma_bf16(a, bcur, acc[ct]);
        if (ct < 7) bcur = bnext;
      }
    }
  }
  int lane = threadIdx.x & 31;
  float* dst = xspec + (long)b * 128 * 128;
#pragma unroll
  for (int ct = 0; ct < 8; ++ct) {
    int c = (ct << 4) + (lane & 15);
#pragma unroll
    for (int j = 0; j < 8; ++j) {
      int k = mbase + ((lane >> 4) << 3) + j;
      unsafeAtomicAdd(&dst[k * 128 + c], acc[ct][j]);
    }
  }
}

// ---------------------------------------------------------------------------
// Fused learned-rotation + tanh:
//   Breal = gX@Are - gY@Aim ; Bimag = gY@Are + gX@Aim
//   gf = tanh(gX*Breal + gY*Bimag)
// Sign of the gY@Aim term handled by flipping bf16 sign bits of the A frag.
// Each B fragment feeds both accumulator sets; fetch pipelined.
// ---------------------------------------------------------------------------
__global__ __launch_bounds__(256) void rot_tanh_kernel(
    const float* __restrict__ gX, const float* __restrict__ gY,
    const float* __restrict__ Are, const float* __restrict__ Aim,
    float* __restrict__ gf, int Mb)
{
  extern __shared__ __align__(16) char smem[];
  __bf16* Xs = (__bf16*)smem;        // gX tile [r][k]
  __bf16* Ys = Xs + 128 * LDP;       // gY tile [r][k]
  __bf16* Rs = Ys + 128 * LDP;       // Are^T   [c][k]
  __bf16* Is = Rs + 128 * LDP;       // Aim^T   [c][k]
  int row0 = blockIdx.x * 128;
  int mbase = (threadIdx.x >> 5) << 4;
  for (int l = threadIdx.x; l < 128 * 128; l += 256) {
    int r = l >> 7, k = l & 127;
    int row = row0 + r;
    bool ok = row < Mb;
    Xs[r * LDP + k] = (__bf16)(ok ? gX[(long)row * 128 + k] : 0.f);
    Ys[r * LDP + k] = (__bf16)(ok ? gY[(long)row * 128 + k] : 0.f);
    Rs[(l & 127) * LDP + (l >> 7)] = (__bf16)Are[l];
    Is[(l & 127) * LDP + (l >> 7)] = (__bf16)Aim[l];
  }
  __syncthreads();
  v8f ar[8] = {}, ai[8] = {};
#pragma unroll
  for (int kb = 0; kb < 4; ++kb) {
    v16bf ax  = frag_A(Xs, mbase, kb);
    v16bf ay  = frag_A(Ys, mbase, kb);
    v16bf ayn = neg_frag(ay);
    v16bf br = frag_B(Rs, 0, kb);
    v16bf bi = frag_B(Is, 0, kb);
#pragma unroll
    for (int ct = 0; ct < 8; ++ct) {
      v16bf brn, bin;
      if (ct < 7) { brn = frag_B(Rs, ct + 1, kb); bin = frag_B(Is, ct + 1, kb); }
      ar[ct] = wmma_bf16(ax,  br, ar[ct]);   // + gX@Are
      ar[ct] = wmma_bf16(ayn, bi, ar[ct]);   // - gY@Aim
      ai[ct] = wmma_bf16(ay,  br, ai[ct]);   // + gY@Are
      ai[ct] = wmma_bf16(ax,  bi, ai[ct]);   // + gX@Aim
      if (ct < 7) { br = brn; bi = bin; }
    }
  }
  int lane = threadIdx.x & 31;
#pragma unroll
  for (int ct = 0; ct < 8; ++ct) {
    int col = (ct << 4) + (lane & 15);
#pragma unroll
    for (int j = 0; j < 8; ++j) {
      int rl  = mbase + ((lane >> 4) << 3) + j;
      int row = row0 + rl;
      if (row < Mb) {
        float gx = (float)Xs[rl * LDP + col];
        float gy = (float)Ys[rl * LDP + col];
        gf[(long)row * 128 + col] = tanhf(gx * ar[ct][j] + gy * ai[ct][j]);
      }
    }
  }
}

// ---------------------------------------------------------------------------
// Small / irregular kernels
// ---------------------------------------------------------------------------
__global__ void zero_kernel(float* p, long n) {
  long i = (long)blockIdx.x * 256 + threadIdx.x;
  if (i < n) p[i] = 0.f;
}

__global__ void first_kernel(const float* __restrict__ x_in,
                             const float* __restrict__ w,
                             const float* __restrict__ bias,
                             float* __restrict__ x, long M) {
  long idx = (long)blockIdx.x * 256 + threadIdx.x;   // M*128
  if (idx >= M * 128) return;
  long row = idx >> 7; int c = idx & 127;
  float s = bias[c];
#pragma unroll
  for (int j = 0; j < 6; ++j) s += x_in[row * 6 + j] * w[j * 128 + c];
  x[idx] = s;
}

__global__ void coef_kernel(const float* __restrict__ evals,
                            const float* __restrict__ dtime,
                            const float* __restrict__ xs,
                            float* __restrict__ s2, int total) {
  int idx = blockIdx.x * 256 + threadIdx.x;
  if (idx >= total) return;
  int c = idx & 127;
  int k = (idx >> 7) & 127;
  int b = idx >> 14;
  s2[idx] = __expf(-evals[b * 128 + k] * dtime[c]) * xs[idx];
}

// COO scatter: gX[rows[e]] += gradX[e]*xd[cols[e]], 4 channels per thread.
__global__ void spmm_kernel(const int* __restrict__ rows,
                            const int* __restrict__ cols,
                            const float* __restrict__ gxv,
                            const float* __restrict__ gyv,
                            const float* __restrict__ xd,
                            float* __restrict__ gX, float* __restrict__ gY,
                            int E, int Nn) {
  int t = blockIdx.x * 256 + threadIdx.x;
  int b = blockIdx.y;
  int e = t >> 5;
  if (e >= E) return;
  int c4 = (t & 31) << 2;
  int r = rows[e], c = cols[e];
  float vx = gxv[(long)b * E + e];
  float vy = gyv[(long)b * E + e];
  const float4 xv = *(const float4*)(xd + ((long)b * Nn + c) * 128 + c4);
  float* dx = gX + ((long)b * Nn + r) * 128 + c4;
  float* dy = gY + ((long)b * Nn + r) * 128 + c4;
  unsafeAtomicAdd(dx + 0, vx * xv.x); unsafeAtomicAdd(dx + 1, vx * xv.y);
  unsafeAtomicAdd(dx + 2, vx * xv.z); unsafeAtomicAdd(dx + 3, vx * xv.w);
  unsafeAtomicAdd(dy + 0, vy * xv.x); unsafeAtomicAdd(dy + 1, vy * xv.y);
  unsafeAtomicAdd(dy + 2, vy * xv.z); unsafeAtomicAdd(dy + 3, vy * xv.w);
}

__global__ void last_kernel(const float* __restrict__ x,
                            const float* __restrict__ w,
                            const float* __restrict__ bias,
                            float* __restrict__ out, long M) {
  long row = (long)blockIdx.x * 256 + threadIdx.x;
  if (row >= M) return;
  float s0 = bias[0], s1 = bias[1], s2 = bias[2];
  const float* xr = x + row * 128;
#pragma unroll 8
  for (int k = 0; k < 128; ++k) {
    float xv = xr[k];
    s0 += xv * w[k * 3 + 0];
    s1 += xv * w[k * 3 + 1];
    s2 += xv * w[k * 3 + 2];
  }
  out[row * 3 + 0] = s0; out[row * 3 + 1] = s1; out[row * 3 + 2] = s2;
}

// ---------------------------------------------------------------------------
extern "C" void kernel_launch(void* const* d_in, const int* in_sizes, int n_in,
                              void* d_out, int out_size, void* d_ws, size_t ws_size,
                              hipStream_t stream) {
  (void)in_sizes; (void)n_in; (void)out_size; (void)ws_size;
  const float* x_in      = (const float*)d_in[0];
  const float* mass      = (const float*)d_in[1];
  const float* evals     = (const float*)d_in[2];
  const float* evecs     = (const float*)d_in[3];
  const int*   rows      = (const int*)d_in[4];
  const int*   cols      = (const int*)d_in[5];
  const float* gxv       = (const float*)d_in[6];
  const float* gyv       = (const float*)d_in[7];
  const float* w_first   = (const float*)d_in[8];
  const float* b_first   = (const float*)d_in[9];
  const float* diff_time = (const float*)d_in[10];
  const float* A_re      = (const float*)d_in[11];
  const float* A_im      = (const float*)d_in[12];
  const float* w0        = (const float*)d_in[13];
  const float* b0        = (const float*)d_in[14];
  const float* w1        = (const float*)d_in[15];
  const float* b1        = (const float*)d_in[16];
  const float* w2        = (const float*)d_in[17];
  const float* b2        = (const float*)d_in[18];
  const float* w_last    = (const float*)d_in[19];
  const float* b_last    = (const float*)d_in[20];

  const int  Bn = 4, Nn = 40000, En = 240000, Kk = 128, CW = 128;
  const long Mtot = (long)Bn * Nn;            // 160000
  const long S    = Mtot * CW;                // 20,480,000 floats / buffer

  float* x  = (float*)d_ws;                   // activations
  float* xd = x  + S;                         // diffused
  float* gX = xd + S;                         // gradX  (reused as h)
  float* gY = gX + S;                         // gradY  (reused as h2)
  float* gf = gY + S;                         // grad features
  float* xs = gf + S;                         // x_spec   [B,K,C]
  float* s2 = xs + (long)Bn * Kk * CW;        // coef*spec [B,K,C]
  float* h  = gX;
  float* h2 = gY;

  const size_t shG = (size_t)(256 + 128) * LDP * sizeof(__bf16);  // ~104 KB (gemm)
  const size_t sh2 = (size_t)2 * 128 * LDP * sizeof(__bf16);      // ~70 KB  (proj)
  const size_t sh4 = (size_t)4 * 128 * LDP * sizeof(__bf16);      // ~139 KB (rot)
  dim3 blk(256);
  const int specTot = Bn * Kk * CW;           // 65536
  const int CHUNK = 512;
  const int nchunks = (Nn + CHUNK - 1) / CHUNK;

  // x = x_in @ w_first + b_first
  first_kernel<<<dim3((unsigned)((Mtot * 128 + 255) / 256)), blk, 0, stream>>>(
      x_in, w_first, b_first, x, Mtot);

  for (int i = 0; i < 4; ++i) {
    // ---- LearnedTimeDiffusion (spectral) ----
    zero_kernel<<<dim3((specTot + 255) / 256), blk, 0, stream>>>(xs, specTot);
    spec_proj_kernel<<<dim3(nchunks, Bn), blk, sh2, stream>>>(
        evecs, x, mass, xs, Nn, CHUNK);
    coef_kernel<<<dim3((specTot + 255) / 256), blk, 0, stream>>>(
        evals, diff_time + i * CW, xs, s2, specTot);
    gemm128_kernel<<<dim3((Nn + 255) / 256, Bn), blk, shG, stream>>>(
        evecs, s2, 1.f, nullptr, nullptr, 0.f, nullptr, nullptr, 0.f,
        nullptr, nullptr, xd, Nn,
        (long)Nn * 128, (long)Kk * CW, (long)Nn * 128, 0);

    // ---- sparse gradient operators ----
    zero_kernel<<<dim3((unsigned)((2 * S + 255) / 256)), blk, 0, stream>>>(gX, 2 * S);
    spmm_kernel<<<dim3((En * 32 + 255) / 256, Bn), blk, 0, stream>>>(
        rows, cols, gxv, gyv, xd, gX, gY, En, Nn);

    // ---- rotations + tanh (fused) ----
    rot_tanh_kernel<<<dim3((unsigned)((Mtot + 127) / 128)), blk, sh4, stream>>>(
        gX, gY, A_re + (long)i * CW * CW, A_im + (long)i * CW * CW, gf, (int)Mtot);

    // ---- MiniMLP + residual ----
    const float* W0i = w0 + (long)i * 384 * 128;
    gemm128_kernel<<<dim3((unsigned)((Mtot + 255) / 256)), blk, shG, stream>>>(
        x,  W0i,             1.f,
        xd, W0i + 128 * 128, 1.f,
        gf, W0i + 256 * 128, 1.f,
        b0 + i * CW, nullptr, h, (int)Mtot, 0, 0, 0, 1);
    gemm128_kernel<<<dim3((unsigned)((Mtot + 255) / 256)), blk, shG, stream>>>(
        h, w1 + (long)i * CW * CW, 1.f, nullptr, nullptr, 0.f,
        nullptr, nullptr, 0.f,
        b1 + i * CW, nullptr, h2, (int)Mtot, 0, 0, 0, 1);
    gemm128_kernel<<<dim3((unsigned)((Mtot + 255) / 256)), blk, shG, stream>>>(
        h2, w2 + (long)i * CW * CW, 1.f, nullptr, nullptr, 0.f,
        nullptr, nullptr, 0.f,
        b2 + i * CW, x, x, (int)Mtot, 0, 0, 0, 0);
  }

  // out = x @ w_last + b_last
  last_kernel<<<dim3((unsigned)((Mtot + 255) / 256)), blk, 0, stream>>>(
      x, w_last, b_last, (float*)d_out, Mtot);
}
